// GAT_3702261809873
// MI455X (gfx1250) — matrix operand order repaired and checked by reading, compiled, not run on hardware
//
#include <hip/hip_runtime.h>
#include <math.h>

typedef unsigned short u16;
typedef unsigned int   u32;
typedef __attribute__((ext_vector_type(16))) __bf16 v16bf;
typedef __attribute__((ext_vector_type(8)))  float  v8f;

#define NNODES 50000
#define DEG    16
#define NEDGE  (NNODES * DEG)
#define DIN    256
#define NHEAD  8
#define DH     32
#define HID    256
#define DOUT   64
#define ALPHA  0.2f

__device__ __forceinline__ u16 f2bf(float f) {
    u32 x = __float_as_uint(f);
    u32 r = (x + 0x7FFFu + ((x >> 16) & 1u)) >> 16;   // round-to-nearest-even
    return (u16)r;
}
__device__ __forceinline__ float bf2f(u16 u) {
    return __uint_as_float(((u32)u) << 16);
}

union FragBF { v16bf v; uint4 q[2]; };

// ---------------------------------------------------------------------------
// Kernel 0: convert + transpose weights to bf16, layout Wt[c][k] (K contiguous)
// W0: [H, IN, DH] -> Wt0[c=h*32+e][d] ; W1: [1, HID, OUT] -> Wt1[c][d]
// ---------------------------------------------------------------------------
__global__ void prep_w(const float* __restrict__ W0, const float* __restrict__ W1,
                       u16* __restrict__ Wt0, u16* __restrict__ Wt1) {
    int idx = blockIdx.x * blockDim.x + threadIdx.x;
    if (idx < 256 * 256) {
        int c = idx >> 8, d = idx & 255;
        Wt0[c * 256 + d] = f2bf(W0[(c >> 5) * (DIN * DH) + d * DH + (c & 31)]);
    }
    if (idx < 64 * 256) {
        int c = idx >> 8, d = idx & 255;
        Wt1[c * 256 + d] = f2bf(W1[d * DOUT + c]);
    }
}

// ---------------------------------------------------------------------------
// WMMA GEMM: Hout[N, COLS](bf16) = A[N, 256] @ Wt^T   (K = 256, bf16 math,
// f32 accumulate). Block = 128 threads (4 waves); block tile = 16 rows x 64
// cols, one 16x16 WMMA tile per wave, 8 K-steps of v_wmma_f32_16x16x32_bf16.
// ---------------------------------------------------------------------------
template <bool A_IS_BF16, int COLS>
__global__ __launch_bounds__(128) void gemm_wmma(const void* __restrict__ Aany,
                                                 const u16* __restrict__ Wt,
                                                 u16* __restrict__ Hout) {
    constexpr int STR = 264;                 // LDS row stride (u16 units), 16B-aligned
    __shared__ u16 Xs[16 * STR];
    const int tid  = threadIdx.x;
    const int row0 = blockIdx.x * 16;

    // Stage 16x256 A-strip into LDS as bf16 (coalesced global loads)
    if (A_IS_BF16) {
        const u16* A = (const u16*)Aany;
        for (int idx = tid; idx < 16 * 256; idx += 128) {
            int r = idx >> 8, c = idx & 255;
            Xs[r * STR + c] = A[(row0 + r) * 256 + c];
        }
    } else {
        const float* A = (const float*)Aany;
        for (int idx = tid; idx < 16 * 256; idx += 128) {
            int r = idx >> 8, c = idx & 255;
            Xs[r * STR + c] = f2bf(A[(row0 + r) * 256 + c]);
        }
    }
    __syncthreads();

    const int wave = tid >> 5;
    const int lane = tid & 31;
    const int half = lane >> 4;              // lane 0-15 vs 16-31
    const int mn   = lane & 15;              // A: row m ; B/D: col n
    const int col0 = blockIdx.y * 64 + wave * 16;

    v8f acc = {};
    const u16* Brow = Wt + (col0 + mn) * 256;  // Wt[c][k]: K contiguous per column

    #pragma unroll
    for (int kb = 0; kb < 256; kb += 32) {
        FragBF a, b;
        // A 16x32 bf16 lane layout: elems 0-7 -> K=kb+half*8+0..7,
        //                           elems 8-15 -> K=kb+16+half*8+0..7
        a.q[0] = *(const uint4*)&Xs[mn * STR + kb + half * 8];
        a.q[1] = *(const uint4*)&Xs[mn * STR + kb + 16 + half * 8];
        // B 32x16: lane holds column n, elems 0-7 -> K=kb+half*8+0..7, elems 8-15 -> K=kb+16+half*8+0..7
        b.q[0] = *(const uint4*)(Brow + kb + half * 8);
        b.q[1] = *(const uint4*)(Brow + kb + 16 + half * 8);
        acc = __builtin_amdgcn_wmma_f32_16x16x32_bf16(
            false, a.v, false, b.v, (short)0, acc, false, false);
    }

    // D f32 16x16 layout: VGPR r -> row (half*8 + r), col n = mn
    #pragma unroll
    for (int r = 0; r < 8; r++) {
        int m = half * 8 + r;
        Hout[(row0 + m) * COLS + col0 + mn] = f2bf(acc[r]);
    }
}

// ---------------------------------------------------------------------------
// Layer-0 attention scores: s_src[n][h], s_dst[n][h] from H0 row dot a0
// ---------------------------------------------------------------------------
__global__ void scores0(const u16* __restrict__ H0, const float* __restrict__ a0,
                        float* __restrict__ Ssrc, float* __restrict__ Sdst) {
    int idx = blockIdx.x * blockDim.x + threadIdx.x;
    if (idx >= NNODES * NHEAD) return;
    int n = idx >> 3, h = idx & 7;
    const u16*  row = H0 + n * HID + h * DH;
    const float* av = a0 + h * (2 * DH);
    float ss = 0.f, sd = 0.f;
    #pragma unroll
    for (int j = 0; j < DH; j++) {
        float v = bf2f(row[j]);
        ss += v * av[j];
        sd += v * av[DH + j];
    }
    Ssrc[n * NHEAD + h] = ss;
    Sdst[n * NHEAD + h] = sd;
}

// ---------------------------------------------------------------------------
// Layer-0 softmax-attention + aggregation + ELU -> X1 bf16 [N, 256]
// One block (256 threads) per source node; its 16 edges are contiguous.
// ---------------------------------------------------------------------------
__global__ __launch_bounds__(256) void agg0(const u16* __restrict__ H0,
                                            const int* __restrict__ dst,
                                            const float* __restrict__ Ssrc,
                                            const float* __restrict__ Sdst,
                                            u16* __restrict__ X1) {
    __shared__ int   ds[DEG];
    __shared__ float esc[NHEAD][DEG];
    __shared__ float att[NHEAD][DEG];
    const int i = blockIdx.x;
    const int t = threadIdx.x;

    if (t < DEG) ds[t] = dst[i * DEG + t];
    __syncthreads();

    if (t < NHEAD * DEG) {
        int h = t >> 4, k = t & 15;
        float e = Ssrc[i * NHEAD + h] + Sdst[ds[k] * NHEAD + h];
        esc[h][k] = e > 0.f ? e : ALPHA * e;           // leaky relu
    }
    __syncthreads();

    if (t < NHEAD) {                                   // stable softmax per head
        float m = -INFINITY;
        for (int k = 0; k < DEG; k++) m = fmaxf(m, esc[t][k]);
        float ex[DEG], s = 0.f;
        for (int k = 0; k < DEG; k++) { ex[k] = expf(esc[t][k] - m); s += ex[k]; }
        float inv = 1.f / s;
        for (int k = 0; k < DEG; k++) att[t][k] = ex[k] * inv;
    }
    __syncthreads();

    const int h = t >> 5;                              // column t -> head t/32
    float a = 0.f;
    #pragma unroll
    for (int k = 0; k < DEG; k++)                      // coalesced row gathers
        a += att[h][k] * bf2f(H0[ds[k] * HID + t]);
    float o = a > 0.f ? a : (expf(a) - 1.f);           // ELU
    X1[i * HID + t] = f2bf(o);
}

// ---------------------------------------------------------------------------
// Layer-1 scores (single head): H1[n,:64] dot a1 halves
// ---------------------------------------------------------------------------
__global__ void scores1(const u16* __restrict__ H1, const float* __restrict__ a1,
                        float* __restrict__ Ssrc, float* __restrict__ Sdst) {
    int n = blockIdx.x * blockDim.x + threadIdx.x;
    if (n >= NNODES) return;
    const u16* row = H1 + n * DOUT;
    float ss = 0.f, sd = 0.f;
    #pragma unroll
    for (int j = 0; j < DOUT; j++) {
        float v = bf2f(row[j]);
        ss += v * a1[j];
        sd += v * a1[DOUT + j];
    }
    Ssrc[n] = ss;
    Sdst[n] = sd;
}

// ---------------------------------------------------------------------------
// Layer-1 attention + aggregation + ELU -> out f32 [N, 64]
// ---------------------------------------------------------------------------
__global__ __launch_bounds__(64) void agg1(const u16* __restrict__ H1,
                                           const int* __restrict__ dst,
                                           const float* __restrict__ Ssrc,
                                           const float* __restrict__ Sdst,
                                           float* __restrict__ out) {
    __shared__ int   ds[DEG];
    __shared__ float att[DEG];
    const int i = blockIdx.x;
    const int t = threadIdx.x;

    if (t < DEG) {
        int d = dst[i * DEG + t];
        ds[t] = d;
        float e = Ssrc[i] + Sdst[d];
        att[t] = e > 0.f ? e : ALPHA * e;
    }
    __syncthreads();
    if (t == 0) {
        float m = -INFINITY;
        for (int k = 0; k < DEG; k++) m = fmaxf(m, att[k]);
        float ex[DEG], s = 0.f;
        for (int k = 0; k < DEG; k++) { ex[k] = expf(att[k] - m); s += ex[k]; }
        float inv = 1.f / s;
        for (int k = 0; k < DEG; k++) att[k] = ex[k] * inv;
    }
    __syncthreads();

    float a = 0.f;
    #pragma unroll
    for (int k = 0; k < DEG; k++)
        a += att[k] * bf2f(H1[ds[k] * DOUT + t]);
    out[i * DOUT + t] = a > 0.f ? a : (expf(a) - 1.f);
}

// ---------------------------------------------------------------------------
extern "C" void kernel_launch(void* const* d_in, const int* in_sizes, int n_in,
                              void* d_out, int out_size, void* d_ws, size_t ws_size,
                              hipStream_t stream) {
    const float* X     = (const float*)d_in[0];
    const int*   edges = (const int*)  d_in[1];
    const float* W0    = (const float*)d_in[2];
    const float* a0    = (const float*)d_in[3];
    const float* W1    = (const float*)d_in[4];
    const float* a1    = (const float*)d_in[5];
    const int*   dst   = edges + NEDGE;            // edges = [src; dst], src implicit
    float*       out   = (float*)d_out;

    // Workspace carving (256-byte aligned slices)
    char* p = (char*)d_ws;
    auto take = [&](size_t bytes) { char* r = p; p += (bytes + 255) & ~(size_t)255; return r; };
    u16*   Wt0  = (u16*)  take((size_t)256 * 256 * 2);
    u16*   Wt1  = (u16*)  take((size_t)64 * 256 * 2);
    u16*   H0   = (u16*)  take((size_t)NNODES * HID * 2);
    u16*   X1   = (u16*)  take((size_t)NNODES * HID * 2);
    u16*   H1   = (u16*)  take((size_t)NNODES * DOUT * 2);
    float* S0s  = (float*)take((size_t)NNODES * NHEAD * 4);
    float* S0d  = (float*)take((size_t)NNODES * NHEAD * 4);
    float* S1s  = (float*)take((size_t)NNODES * 4);
    float* S1d  = (float*)take((size_t)NNODES * 4);

    prep_w<<<(256 * 256 + 255) / 256, 256, 0, stream>>>(W0, W1, Wt0, Wt1);

    // Layer 0: H0 = X @ W0 (bf16 WMMA, f32 acc)
    gemm_wmma<false, HID><<<dim3(NNODES / 16, HID / 64), 128, 0, stream>>>(X, Wt0, H0);
    scores0<<<(NNODES * NHEAD + 255) / 256, 256, 0, stream>>>(H0, a0, S0s, S0d);
    agg0<<<NNODES, 256, 0, stream>>>(H0, dst, S0s, S0d, X1);

    // Layer 1: H1 = X1 @ W1
    gemm_wmma<true, DOUT><<<dim3(NNODES / 16, 1), 128, 0, stream>>>(X1, Wt1, H1);
    scores1<<<(NNODES + 255) / 256, 256, 0, stream>>>(H1, a1, S1s, S1d);
    agg1<<<NNODES, 64, 0, stream>>>(H1, dst, S1s, S1d, out);
}
